// LSTMModel_42434276884914
// MI455X (gfx1250) — compile-verified, matrix operand order checked by reading, not run on hardware
//
#include <hip/hip_runtime.h>
#include <hip/hip_bf16.h>

typedef __attribute__((ext_vector_type(16))) __bf16 v16bf;
typedef __attribute__((ext_vector_type(8)))  float  v8f;

#define BATCH 256
#define TSTEPS 512
#define IDIM 64
#define HDIM 512
#define GDIM 2048   // 4*H

// ---------------------------------------------------------------------------
// Weight packing: f32 -> bf16, WMMA B-fragment layout.
// Packed flat index = ((nt*KT + kt)*32 + lane)*16 + hh
// B-frag mapping: lane l -> N = l%16 ; half hh -> K = kt*32 + (l<16 ? hh : 16+hh)
// Byte address = nt*(KT*1024) + kt*1024 + lane*32 + hh*2
// ---------------------------------------------------------------------------
__global__ void pack_whh(const float* __restrict__ whh, __hip_bfloat16* __restrict__ out) {
    int idx = blockIdx.x * blockDim.x + threadIdx.x;   // 0 .. 2048*512-1
    if (idx >= GDIM * HDIM) return;
    int hh   =  idx        & 15;
    int lane = (idx >> 4)  & 31;
    int kt   = (idx >> 9)  & 15;   // 16 K-tiles (K = 512)
    int nt   =  idx >> 13;         // 128 N-tiles
    int n = nt * 16 + (lane & 15);
    int k = kt * 32 + ((lane < 16) ? hh : 16 + hh);
    out[idx] = __float2bfloat16(whh[n * HDIM + k]);
}

__global__ void pack_wih(const float* __restrict__ wih, __hip_bfloat16* __restrict__ out) {
    int idx = blockIdx.x * blockDim.x + threadIdx.x;   // 0 .. 2048*64-1
    if (idx >= GDIM * IDIM) return;
    int hh   =  idx        & 15;
    int lane = (idx >> 4)  & 31;
    int kt   = (idx >> 9)  & 1;    // 2 K-tiles (K = 64)
    int nt   =  idx >> 10;         // 128 N-tiles
    int n = nt * 16 + (lane & 15);
    int k = kt * 32 + ((lane < 16) ? hh : 16 + hh);
    out[idx] = __float2bfloat16(wih[n * IDIM + k]);
}

// ---------------------------------------------------------------------------
__device__ __forceinline__ float fsigmoid(float x) {
    return 1.0f / (1.0f + __expf(-x));
}
__device__ __forceinline__ float ftanh(float x) {
    float e = __expf(2.0f * x);
    return (e - 1.0f) / (e + 1.0f);
}
__device__ __forceinline__ v8f splat8(float v) {
    v8f r;
#pragma unroll
    for (int e = 0; e < 8; ++e) r[e] = v;
    return r;
}

// ---------------------------------------------------------------------------
// Persistent LSTM kernel. 16 workgroups x 512 threads (16 waves of 32).
// Workgroup owns 16 batch rows; wave owns hidden slices nt = g*32 + (w + 16r),
// r=0..1, for all 4 gates -> nonlinear cell update is register-local.
// Acc footprint per wave: 8 x v8f = 64 VGPRs; cell state 16 VGPRs.
// K-tile loop kept rolled so only ~2 B-fragments are in flight (no spills).
// ---------------------------------------------------------------------------
__global__ __launch_bounds__(512, 1)
void lstm_persistent(const float* __restrict__ x,
                     const __hip_bfloat16* __restrict__ wih_p,
                     const __hip_bfloat16* __restrict__ whh_p,
                     const float* __restrict__ b_ih,
                     const float* __restrict__ b_hh,
                     const float* __restrict__ w_fc,
                     const float* __restrict__ b_fc,
                     float* __restrict__ out) {
    __shared__ __align__(32) __hip_bfloat16 h_lds[16 * HDIM];   // 16 KB, A-operand feed
    __shared__ __align__(32) __hip_bfloat16 x_lds[16 * IDIM];   // 2 KB, staged x_t

    const int tid = threadIdx.x;
    const int w   = tid >> 5;          // wave 0..15
    const int l   = tid & 31;          // lane 0..31
    const int wg  = blockIdx.x;        // batch rows [wg*16, wg*16+16)

    // h_0 = 0
    for (int i = tid; i < 16 * HDIM; i += 512) h_lds[i] = __float2bfloat16(0.0f);

    // Bias preload: in C/D layout every lane holds a fixed N = l%16 per tile.
    float bias[4][2];
#pragma unroll
    for (int g = 0; g < 4; ++g)
#pragma unroll
        for (int r = 0; r < 2; ++r) {
            int n = g * 512 + (w + 16 * r) * 16 + (l & 15);
            bias[g][r] = b_ih[n] + b_hh[n];
        }

    // Persistent cell state in accumulator layout (2 hidden tiles per wave).
    v8f c[2];
#pragma unroll
    for (int r = 0; r < 2; ++r) c[r] = splat8(0.0f);

    const int am     = l & 15;                 // A-frag M row
    const int ak_off = (l < 16) ? 0 : 16;      // A-frag K sub-offset (halves)

    // Per-wave/lane base pointers; everything else is an immediate offset.
    const char* whh_base0 = (const char*)whh_p + (w * 16384 + l * 32);
    const char* wih_base  = (const char*)wih_p + (w * 2048  + l * 32);
    const __hip_bfloat16* ha_base = &h_lds[am * HDIM + ak_off];
    const __hip_bfloat16* xa_base = &x_lds[am * IDIM + ak_off];
    // h writeback: lane stores K fixed per r, M = e + 8*(l>>4)
    __hip_bfloat16* hw_base = &h_lds[(8 * (l >> 4)) * HDIM + w * 16 + (l & 15)];

    const float* x_base = x + ((size_t)(wg * 16) * TSTEPS) * IDIM;

    __syncthreads();

    for (int t = 0; t < TSTEPS; ++t) {
        // ---- stage x[rows, t, :] into LDS as bf16 (16x64) ----
        {
            int idx = tid * 2;                 // 16*64/512 = 2 elems per thread
            int m  = idx >> 6;
            int i0 = idx & 63;
            const float2 v = *(const float2*)(x_base + ((size_t)m * TSTEPS + t) * IDIM + i0);
            x_lds[m * IDIM + i0 + 0] = __float2bfloat16(v.x);
            x_lds[m * IDIM + i0 + 1] = __float2bfloat16(v.y);
        }
        __syncthreads();

        // ---- gates = bias + h@Whh^T + x@Wih^T  (WMMA bf16, f32 accum) ----
        v8f acc[4][2];
#pragma unroll
        for (int g = 0; g < 4; ++g)
#pragma unroll
            for (int r = 0; r < 2; ++r) acc[g][r] = splat8(bias[g][r]);

        // recurrent part: K = 512 -> 16 K-tiles; rolled loop, one pointer bump
        // per tile, fragment loads at compile-time immediate offsets.
        {
            const char* bp = whh_base0;
#pragma unroll 1
            for (int kt = 0; kt < 16; ++kt) {
                v16bf a = *(const v16bf*)(ha_base + kt * 32);
                __builtin_prefetch(bp + 1024, 0, 1);   // next K-tile
#pragma unroll
                for (int g = 0; g < 4; ++g)
#pragma unroll
                    for (int r = 0; r < 2; ++r) {
                        v16bf b = *(const v16bf*)(bp + g * 524288 + r * 262144);
                        acc[g][r] = __builtin_amdgcn_wmma_f32_16x16x32_bf16(
                            false, a, false, b, (short)0, acc[g][r], false, false);
                    }
                bp += 1024;
            }
        }
        // input part: K = 64 -> 2 K-tiles
#pragma unroll 1
        for (int kt = 0; kt < 2; ++kt) {
            v16bf a = *(const v16bf*)(xa_base + kt * 32);
#pragma unroll
            for (int g = 0; g < 4; ++g)
#pragma unroll
                for (int r = 0; r < 2; ++r) {
                    v16bf b = *(const v16bf*)(wih_base + kt * 1024 + g * 65536 + r * 32768);
                    acc[g][r] = __builtin_amdgcn_wmma_f32_16x16x32_bf16(
                        false, a, false, b, (short)0, acc[g][r], false, false);
                }
        }

        // all waves done reading h_lds before it is overwritten
        __syncthreads();

        // ---- cell update (register-local) + h writeback in A-feed layout ----
#pragma unroll
        for (int r = 0; r < 2; ++r) {
            __hip_bfloat16* hw = hw_base + r * 256;     // K += 16r*16
#pragma unroll
            for (int e = 0; e < 8; ++e) {
                float ig = fsigmoid(acc[0][r][e]);
                float fg = fsigmoid(acc[1][r][e]);
                float gg = ftanh   (acc[2][r][e]);
                float og = fsigmoid(acc[3][r][e]);
                float cc = fg * c[r][e] + ig * gg;
                c[r][e] = cc;
                hw[e * HDIM] = __float2bfloat16(og * ftanh(cc));
            }
        }
        __syncthreads();
    }

    // ---- final FC: out[b] = h_T[b,:] . W_fc + b_fc ----
    if (tid < 16) {
        float s = b_fc[0];
        for (int k = 0; k < HDIM; ++k)
            s += __bfloat162float(h_lds[tid * HDIM + k]) * w_fc[k];
        out[wg * 16 + tid] = s;
    }
}

// ---------------------------------------------------------------------------
extern "C" void kernel_launch(void* const* d_in, const int* in_sizes, int n_in,
                              void* d_out, int out_size, void* d_ws, size_t ws_size,
                              hipStream_t stream) {
    const float* x    = (const float*)d_in[0];
    const float* W_ih = (const float*)d_in[1];
    const float* W_hh = (const float*)d_in[2];
    const float* b_ih = (const float*)d_in[3];
    const float* b_hh = (const float*)d_in[4];
    const float* W_fc = (const float*)d_in[5];
    const float* b_fc = (const float*)d_in[6];

    __hip_bfloat16* whh_p = (__hip_bfloat16*)d_ws;                      // 2 MB
    __hip_bfloat16* wih_p = whh_p + (size_t)GDIM * HDIM;                // +256 KB

    pack_whh<<<(GDIM * HDIM) / 256, 256, 0, stream>>>(W_hh, whh_p);
    pack_wih<<<(GDIM * IDIM) / 256, 256, 0, stream>>>(W_ih, wih_p);

    lstm_persistent<<<BATCH / 16, 512, 0, stream>>>(
        x, wih_p, whh_p, b_ih, b_hh, W_fc, b_fc, (float*)d_out);
}
